// RNN_60352880443473
// MI455X (gfx1250) — compile-verified, matrix operand order checked by reading, not run on hardware
//
#include <hip/hip_runtime.h>
#include <math.h>

#define B_  512
#define S_  512
#define D_  128
#define H_  256
#define G_  1024   // 4*H

#if defined(__has_builtin)
# if __has_builtin(__builtin_amdgcn_global_load_async_to_lds_b128) && \
     __has_builtin(__builtin_amdgcn_s_wait_asynccnt)
#  define HAVE_ASYNC 1
# endif
#endif
#ifndef HAVE_ASYNC
# define HAVE_ASYNC 0
#endif

typedef __attribute__((ext_vector_type(16))) __bf16 v16bf;
typedef __attribute__((ext_vector_type(8)))  float  v8f;
typedef __attribute__((ext_vector_type(4)))  unsigned u32x4;
typedef __attribute__((ext_vector_type(4)))  float    f32x4;
typedef __attribute__((ext_vector_type(4)))  int      i32x4;

__device__ __forceinline__ unsigned short f2bf(float f) {
    union { float f; unsigned u; } x; x.f = f;
    unsigned r = x.u + 0x7fffu + ((x.u >> 16) & 1u);   // round-to-nearest-even
    return (unsigned short)(r >> 16);
}
__device__ __forceinline__ float bf2f(unsigned short h) {
    union { unsigned u; float f; } x; x.u = ((unsigned)h) << 16;
    return x.f;
}

#if HAVE_ASYNC
// GLOBAL_LOAD_ASYNC_TO_LDS_B128: moves 16B/lane global->LDS, tracked by ASYNCcnt.
// Builtin signature (from hipcc diagnostic): (int4 AS(1)*, int4 AS(3)*, Imm off, Imm cpol)
__device__ __forceinline__ void async_copy16(const void* g, void* l) {
    __builtin_amdgcn_global_load_async_to_lds_b128(
        (__attribute__((address_space(1))) i32x4*)g,
        (__attribute__((address_space(3))) i32x4*)l, 0, 0);
}
#endif

// ---------------------------------------------------------------------------
// Repack a row-major fp32 weight [K x 1024] into bf16 WMMA B-fragments
// (32x16 tiles, ISA 7.12.2 16-bit layout). frag = ntile*ksteps + ks, 1KB/frag.
// ---------------------------------------------------------------------------
__global__ void repack_w(const float* __restrict__ src, unsigned* __restrict__ dst,
                         int ksteps) {
    int tid  = blockIdx.x * blockDim.x + threadIdx.x;
    int frag = tid >> 8;
    int rem  = tid & 255;
    int lane = rem >> 3;
    int d    = rem & 7;
    int ntile = frag / ksteps;
    int ks    = frag - ntile * ksteps;
    int klocal = 2 * (d & 3) + ((d >= 4) ? 16 : 0) + ((lane >= 16) ? 8 : 0);
    int k = ks * 32 + klocal;
    int n = ntile * 16 + (lane & 15);
    unsigned short lo = f2bf(src[(size_t)k * G_ + n]);
    unsigned short hi = f2bf(src[(size_t)(k + 1) * G_ + n]);
    dst[(size_t)frag * 256 + lane * 8 + d] = (unsigned)lo | ((unsigned)hi << 16);
}

// ---------------------------------------------------------------------------
// xz[t][btile] = bf16(x_tile @ Wx + b), stored as packed bf16 C-fragments:
// frag index (t*32+btile)*64 + ntile, 512B/frag (lane holds 8 bf16 = 16B).
// Throughput kernel: 16384 blocks hide L2 latency via occupancy.
// ---------------------------------------------------------------------------
__global__ __launch_bounds__(256) void gemm_xz(
    const float* __restrict__ x,       // [B][S][D] fp32
    const u32x4* __restrict__ wxp,     // packed Wx B-frags (64 ntiles x 4 ksteps)
    const float* __restrict__ bias,    // [G]
    u32x4*       __restrict__ xzp)     // packed xz C-frags
{
    __shared__ __align__(16) unsigned short xl[16 * 128]; // bf16 A tile (4KB)
    int blk   = blockIdx.x;            // t*32 + btile
    int t     = blk >> 5;
    int btile = blk & 31;
    int tid   = threadIdx.x;

    { // stage x tile -> bf16 LDS, row-major (matches A-fragment lane layout)
        int m  = tid >> 4;
        int ch = tid & 15;
        const float* row = x + ((size_t)(btile * 16 + m) * S_ + t) * D_ + ch * 8;
        f32x4 f0 = *(const f32x4*)(row);
        f32x4 f1 = *(const f32x4*)(row + 4);
        unsigned short* o = xl + m * 128 + ch * 8;
        o[0]=f2bf(f0.x); o[1]=f2bf(f0.y); o[2]=f2bf(f0.z); o[3]=f2bf(f0.w);
        o[4]=f2bf(f1.x); o[5]=f2bf(f1.y); o[6]=f2bf(f1.z); o[7]=f2bf(f1.w);
    }
    __syncthreads();

    int w = tid >> 5, lane = tid & 31;
    int rowl = lane & 15, hi = (lane >> 4) & 1;
    const u32x4* wbase = wxp + lane * 2;   // + (ng*4+ks)*64

    v8f acc[8];
    #pragma unroll
    for (int nn = 0; nn < 8; ++nn) {
        float bv = bias[(w * 8 + nn) * 16 + rowl];
        #pragma unroll
        for (int j = 0; j < 8; ++j) acc[nn][j] = bv;
    }

    #pragma unroll
    for (int ks = 0; ks < 4; ++ks) {
        union { u32x4 q[2]; v16bf v; } a;
        const u32x4* ap = (const u32x4*)(xl + rowl * 128 + ks * 32 + hi * 8);
        a.q[0] = ap[0];
        a.q[1] = ap[2];
        #pragma unroll
        for (int nn = 0; nn < 8; ++nn) {
            const u32x4* bp = wbase + (size_t)((w * 8 + nn) * 4 + ks) * 64;
            union { u32x4 q[2]; v16bf v; } b;
            b.q[0] = bp[0]; b.q[1] = bp[1];
            acc[nn] = __builtin_amdgcn_wmma_f32_16x16x32_bf16(
                false, a.v, false, b.v, (short)0, acc[nn], false, false);
        }
    }

    size_t fbase = (size_t)blk * 64;
    #pragma unroll
    for (int nn = 0; nn < 8; ++nn) {
        int ng = w * 8 + nn;
        u32x4 o;
        o.x = (unsigned)f2bf(acc[nn][0]) | ((unsigned)f2bf(acc[nn][1]) << 16);
        o.y = (unsigned)f2bf(acc[nn][2]) | ((unsigned)f2bf(acc[nn][3]) << 16);
        o.z = (unsigned)f2bf(acc[nn][4]) | ((unsigned)f2bf(acc[nn][5]) << 16);
        o.w = (unsigned)f2bf(acc[nn][6]) | ((unsigned)f2bf(acc[nn][7]) << 16);
        xzp[(fbase + ng) * 32 + lane] = o;
    }
}

// ---------------------------------------------------------------------------
// Persistent LSTM scan: 32 blocks x 16 batch rows, 8 waves. h,c live in LDS.
// Wh B-fragments are streamed L2 -> LDS with GLOBAL_LOAD_ASYNC_TO_LDS_B128
// into a per-K-step double buffer; s_wait_asynccnt gates consumption so the
// async engine always runs one K-step (16 ops) ahead of the WMMA pipe.
// ---------------------------------------------------------------------------
__global__ __launch_bounds__(256) void lstm_scan(
    const u32x4* __restrict__ xzp,   // packed xz C-frags (bf16)
    const u32x4* __restrict__ whp,   // packed Wh B-frags (64 ntiles x 8 ksteps)
    float*       __restrict__ hfin)  // [B][H] fp32 final hidden state
{
    __shared__ __align__(16) float          zl[16 * G_];  // 64KB gate pre-acts
    __shared__ __align__(16) float          cl[16 * H_];  // 16KB cell state
    __shared__ __align__(16) unsigned short hl[16 * H_];  //  8KB h (bf16, A layout)
#if HAVE_ASYNC
    __shared__ __align__(16) unsigned short whl[2][8 * 4096]; // 2x64KB Wh staging
#endif

    int btile = blockIdx.x;
    int tid   = threadIdx.x;
    for (int i = tid; i < 16 * H_; i += 256) { cl[i] = 0.f; hl[i] = 0; }
    __syncthreads();

    int w = tid >> 5, lane = tid & 31;
    int rowl = lane & 15, hi = (lane >> 4) & 1;
    int gr = tid >> 4;            // gate row 0..15
    int gc0 = (tid & 15) * 16;    // gate col base
    const u32x4* wbase = whp + lane * 2;   // + (ng*8+ks)*64  (fallback path)

#if HAVE_ASYNC
    // Issue one K-step batch (8 frags = 16 async b128 ops) for this wave.
    #define ISSUE_WH_BATCH(bufp, ksv)                                          \
    {                                                                          \
        _Pragma("unroll")                                                      \
        for (int nn = 0; nn < 8; ++nn) {                                       \
            const char* g = (const char*)whp +                                 \
                (size_t)((w * 8 + nn) * 8 + (ksv)) * 1024 + lane * 32;         \
            char* l = (char*)&whl[(bufp)][w * 4096] + nn * 1024 + lane * 32;   \
            async_copy16(g, l);                                                \
            async_copy16(g + 16, l + 16);                                      \
        }                                                                      \
    }
    ISSUE_WH_BATCH(0, 0)   // prologue: buffer 0 <- ks=0 of step 0
#endif

    for (int t = 0; t < S_; ++t) {
        // --- xz fragment loads (direct, LOADcnt) ---
        u32x4 xq[8];
        size_t fbase = ((size_t)t * 32 + btile) * 64;
        const u32x4* xzbase = xzp + (fbase + w * 8) * 32 + lane;
        #pragma unroll
        for (int nn = 0; nn < 8; ++nn) xq[nn] = xzbase[nn * 32];

        // --- unpack xz (bf16 C-fragments) into fp32 accumulators ---
        v8f acc[8];
        #pragma unroll
        for (int nn = 0; nn < 8; ++nn) {
            unsigned qq[4] = {xq[nn].x, xq[nn].y, xq[nn].z, xq[nn].w};
            #pragma unroll
            for (int j = 0; j < 4; ++j) {
                acc[nn][2*j]   = bf2f((unsigned short)(qq[j] & 0xffffu));
                acc[nn][2*j+1] = bf2f((unsigned short)(qq[j] >> 16));
            }
        }
        if (t + 1 < S_) { // global_prefetch_b8 for next step's xz stream
            const char* nf = (const char*)(xzp +
                (((size_t)(t + 1) * 32 + btile) * 64 + w * 8) * 32 + lane);
            __builtin_prefetch(nf, 0, 1);
        }

        // --- z += h @ Wh : 8 K-steps x 8 N-tiles ---
        #pragma unroll
        for (int ks = 0; ks < 8; ++ks) {
            int cur = ks & 1, nxt = cur ^ 1;
#if HAVE_ASYNC
            // keep the async engine one K-step ahead; at ks==7, pre-issue the
            // next timestep's ks=0 so gates+barriers hide its latency.
            if (ks < 7) {
                ISSUE_WH_BATCH(nxt, ks + 1)
                __builtin_amdgcn_s_wait_asynccnt(16);
            } else if (t + 1 < S_) {
                ISSUE_WH_BATCH(nxt, 0)
                __builtin_amdgcn_s_wait_asynccnt(16);
            } else {
                __builtin_amdgcn_s_wait_asynccnt(0);
            }
            asm volatile("" ::: "memory");  // don't hoist ds reads above wait
#endif
            union { u32x4 q[2]; v16bf v; } a;
            const u32x4* ap = (const u32x4*)(hl + rowl * H_ + ks * 32 + hi * 8);
            a.q[0] = ap[0];
            a.q[1] = ap[2];
            #pragma unroll
            for (int nn = 0; nn < 8; ++nn) {
                union { u32x4 q[2]; v16bf v; } b;
#if HAVE_ASYNC
                const u32x4* bsrc = (const u32x4*)
                    ((const char*)&whl[cur][w * 4096] + nn * 1024 + lane * 32);
                b.q[0] = bsrc[0]; b.q[1] = bsrc[1];
#else
                const u32x4* bp = wbase + (size_t)((w * 8 + nn) * 8 + ks) * 64;
                b.q[0] = bp[0]; b.q[1] = bp[1];
#endif
                acc[nn] = __builtin_amdgcn_wmma_f32_16x16x32_bf16(
                    false, a.v, false, b.v, (short)0, acc[nn], false, false);
            }
        }

        // --- scatter z fragments to LDS ---
        #pragma unroll
        for (int nn = 0; nn < 8; ++nn) {
            int n = (w * 8 + nn) * 16 + rowl;
            #pragma unroll
            for (int j = 0; j < 8; ++j)
                zl[(j + hi * 8) * G_ + n] = acc[nn][j];
        }
        __syncthreads();

        // --- gates: i,f,g,o -> update c (fp32) and h (bf16) in LDS ---
        #pragma unroll 4
        for (int cc = 0; cc < 16; ++cc) {
            int c = gc0 + cc;
            float zi = zl[gr * G_ + c];
            float zf = zl[gr * G_ + H_  + c];
            float zg = zl[gr * G_ + 2*H_ + c];
            float zo = zl[gr * G_ + 3*H_ + c];
            float ig = 1.f / (1.f + __expf(-zi));
            float fg = 1.f / (1.f + __expf(-zf));
            float og = 1.f / (1.f + __expf(-zo));
            float gg = tanhf(zg);
            float cn = fg * cl[gr * H_ + c] + ig * gg;
            float hn = og * tanhf(cn);
            cl[gr * H_ + c] = cn;
            hl[gr * H_ + c] = f2bf(hn);
            if (t == S_ - 1)
                hfin[(size_t)(btile * 16 + gr) * H_ + c] = hn;
        }
        __syncthreads();
    }
}

// ---------------------------------------------------------------------------
// Head: logits = h @ Wd + bd, softmax. Tiny (512x256x10).
// ---------------------------------------------------------------------------
__global__ void head_kernel(const float* __restrict__ hfin,
                            const float* __restrict__ Wd,
                            const float* __restrict__ bd,
                            float* __restrict__ out) {
    int row = blockIdx.x * blockDim.x + threadIdx.x;
    if (row >= B_) return;
    float lg[10];
    #pragma unroll
    for (int j = 0; j < 10; ++j) lg[j] = bd[j];
    for (int k = 0; k < H_; ++k) {
        float hv = hfin[(size_t)row * H_ + k];
        #pragma unroll
        for (int j = 0; j < 10; ++j) lg[j] += hv * Wd[k * 10 + j];
    }
    float mx = lg[0];
    #pragma unroll
    for (int j = 1; j < 10; ++j) mx = fmaxf(mx, lg[j]);
    float s = 0.f;
    #pragma unroll
    for (int j = 0; j < 10; ++j) { lg[j] = __expf(lg[j] - mx); s += lg[j]; }
    float inv = 1.f / s;
    #pragma unroll
    for (int j = 0; j < 10; ++j) out[row * 10 + j] = lg[j] * inv;
}

extern "C" void kernel_launch(void* const* d_in, const int* in_sizes, int n_in,
                              void* d_out, int out_size, void* d_ws, size_t ws_size,
                              hipStream_t stream) {
    const float* x  = (const float*)d_in[0];
    const float* Wx = (const float*)d_in[1];
    const float* Wh = (const float*)d_in[2];
    const float* b  = (const float*)d_in[3];
    const float* Wd = (const float*)d_in[4];
    const float* bd = (const float*)d_in[5];
    float* out = (float*)d_out;

    char* ws = (char*)d_ws;
    size_t off = 0;
    u32x4*    xzp = (u32x4*)(ws + off);    off += (size_t)S_ * B_ * G_ * 2; // 512MB bf16 frags
    unsigned* whp = (unsigned*)(ws + off); off += (size_t)H_ * G_ * 2;      // 512KB
    unsigned* wxp = (unsigned*)(ws + off); off += (size_t)D_ * G_ * 2;      // 256KB
    float*   hfin = (float*)(ws + off);    off += (size_t)B_ * H_ * 4;      // 512KB

    // 1) repack weights into bf16 WMMA B-fragment layout (L2-resident)
    repack_w<<<512, 256, 0, stream>>>(Wh, whp, 8);   // 64 ntiles x 8 ksteps
    repack_w<<<256, 256, 0, stream>>>(Wx, wxp, 4);   // 64 ntiles x 4 ksteps
    // 2) parallel input projection -> packed bf16 xz fragments
    gemm_xz<<<S_ * 32, 256, 0, stream>>>(x, (const u32x4*)wxp, b, xzp);
    // 3) persistent per-batch-tile LSTM scan (WMMA + async-LDS Wh stream)
    lstm_scan<<<32, 256, 0, stream>>>(xzp, (const u32x4*)whp, hfin);
    // 4) dense head + softmax
    head_kernel<<<2, 256, 0, stream>>>(hfin, Wd, bd, out);
}
